// TransformerBlock_88785563942916
// MI455X (gfx1250) — matrix operand order repaired and checked by reading, not compiled, not run on hardware
//
#include <hip/hip_runtime.h>

// ---------------------------------------------------------------------------
// Transformer block for MI455X (gfx1250): bf16 WMMA GEMMs + flash attention,
// with async global->LDS double-buffered staging (ASYNCcnt-tracked).
// B=8 T=1024 D=1024 H=16 KVH=8 HD=64, M = B*T = 8192.
// ---------------------------------------------------------------------------

typedef unsigned short u16;
typedef __attribute__((ext_vector_type(16))) __bf16 v16bf;
typedef __attribute__((ext_vector_type(8)))  float  v8f;
typedef int v4i_ __attribute__((vector_size(16)));   // matches builtin's V4i

union FragAB { v16bf v; uint4 q[2]; };   // 32 bytes: one WMMA bf16 A/B operand

#ifndef __has_builtin
#define __has_builtin(x) 0
#endif

#define AS1 __attribute__((address_space(1)))
#define AS3 __attribute__((address_space(3)))

#if __has_builtin(__builtin_amdgcn_global_load_async_to_lds_b128)
#define ASYNC_COPY 1
#else
#define ASYNC_COPY 0
#endif

// 16-byte global -> LDS copy. Async (ASYNCcnt-tracked, no VGPR round trip) on
// gfx1250; plain load+store fallback otherwise.
__device__ __forceinline__ void copy_b128(const u16* g, u16* l) {
#if ASYNC_COPY
  __builtin_amdgcn_global_load_async_to_lds_b128(
      (AS1 v4i_*)(uintptr_t)g,
      (AS3 v4i_*)(unsigned)(uintptr_t)l, 0, 0);
#else
  *(uint4*)l = *(const uint4*)g;
#endif
}

template <int N>
__device__ __forceinline__ void async_wait() {
#if ASYNC_COPY
#if __has_builtin(__builtin_amdgcn_s_wait_asynccnt)
  __builtin_amdgcn_s_wait_asynccnt(N);
#else
  asm volatile("s_wait_asynccnt %0" :: "i"(N) : "memory");
#endif
#endif
}

__device__ __forceinline__ u16 f2bf(float f) {            // RNE float->bf16
  unsigned u = __float_as_uint(f);
  unsigned r = u + 0x7FFFu + ((u >> 16) & 1u);
  return (u16)(r >> 16);
}
__device__ __forceinline__ v8f vzero8() {
  v8f z = {0.f,0.f,0.f,0.f,0.f,0.f,0.f,0.f}; return z;
}

// ---------------------------------------------------------------------------
// Weight transpose + convert: dst[N][K] (bf16) = src[K][N] (f32)
// ---------------------------------------------------------------------------
__global__ __launch_bounds__(256) void transpose_cvt(
    const float* __restrict__ src, u16* __restrict__ dst, int K, int N) {
  size_t id = (size_t)blockIdx.x * 256 + threadIdx.x;
  int n = (int)(id / (size_t)K), k = (int)(id % (size_t)K);
  dst[id] = f2bf(src[(size_t)k * N + n]);
}

// ---------------------------------------------------------------------------
// LayerNorm row kernel: f32 [rows][1024] -> bf16 [rows][1024]
// ---------------------------------------------------------------------------
__global__ __launch_bounds__(256) void layernorm_bf16(
    const float* __restrict__ x, const float* __restrict__ gam,
    const float* __restrict__ bet, u16* __restrict__ out) {
  const int row = blockIdx.x, t = threadIdx.x;
  const float4 xv = *(const float4*)&x[(size_t)row * 1024 + t * 4];
  float s  = xv.x + xv.y + xv.z + xv.w;
  float s2 = xv.x*xv.x + xv.y*xv.y + xv.z*xv.z + xv.w*xv.w;
  #pragma unroll
  for (int off = 1; off < 32; off <<= 1) {
    s  += __shfl_xor(s,  off, 32);
    s2 += __shfl_xor(s2, off, 32);
  }
  __shared__ float rs[8], rs2[8];
  const int w = t >> 5, l = t & 31;
  if (l == 0) { rs[w] = s; rs2[w] = s2; }
  __syncthreads();
  float ts = 0.f, ts2 = 0.f;
  #pragma unroll
  for (int i = 0; i < 8; ++i) { ts += rs[i]; ts2 += rs2[i]; }
  const float mu  = ts * (1.f / 1024.f);
  const float var = ts2 * (1.f / 1024.f) - mu * mu;
  const float inv = rsqrtf(var + 1e-5f);
  const float4 gv = *(const float4*)&gam[t * 4];
  const float4 bv = *(const float4*)&bet[t * 4];
  u16 o0 = f2bf((xv.x - mu) * inv * gv.x + bv.x);
  u16 o1 = f2bf((xv.y - mu) * inv * gv.y + bv.y);
  u16 o2 = f2bf((xv.z - mu) * inv * gv.z + bv.z);
  u16 o3 = f2bf((xv.w - mu) * inv * gv.w + bv.w);
  uint2 pk;
  pk.x = (unsigned)o0 | ((unsigned)o1 << 16);
  pk.y = (unsigned)o2 | ((unsigned)o3 << 16);
  *(uint2*)&out[(size_t)row * 1024 + t * 4] = pk;
}

// ---------------------------------------------------------------------------
// bf16 WMMA GEMM: C[M,N] = A[M,K] @ B[K,N] + bias (B pre-transposed as Bt[N][K])
// Block tile 128x128x32, double-buffered async LDS staging; 8 waves, each wave
// a 32x64 sub-tile (2x4 v_wmma_f32_16x16x32_bf16 per k-step).
// ---------------------------------------------------------------------------
template <bool OUT_BF, bool OUT_F, bool RELU, bool RES>
__global__ __launch_bounds__(256) void gemm_bf16(
    const u16* __restrict__ A, const u16* __restrict__ Bt,
    const float* __restrict__ bias, const float* __restrict__ res,
    u16* __restrict__ outB, float* __restrict__ outF,
    int M, int N, int K) {
  __shared__ __align__(16) u16 As[2][128 * 32];
  __shared__ __align__(16) u16 Bs[2][128 * 32];
  const int n0 = blockIdx.x * 128, m0 = blockIdx.y * 128;
  const int t = threadIdx.x, w = t >> 5, l = t & 31, g = l >> 4, lm = l & 15;
  const int wm = w & 3, wn = w >> 2;     // wave tile: rows wm*32, cols wn*64
  const int r0 = t >> 2, ch0 = (t & 3) * 8;   // staging: rows r0 and r0+64

  v8f acc[2][4];
  #pragma unroll
  for (int i = 0; i < 2; ++i)
    #pragma unroll
    for (int j = 0; j < 4; ++j) acc[i][j] = vzero8();

  auto stage = [&](int buf, int k0) {   // 128x32 tiles of A and Bt
    copy_b128(&A [(size_t)(m0 + r0)      * K + k0 + ch0], &As[buf][ r0       * 32 + ch0]);
    copy_b128(&A [(size_t)(m0 + r0 + 64) * K + k0 + ch0], &As[buf][(r0 + 64) * 32 + ch0]);
    copy_b128(&Bt[(size_t)(n0 + r0)      * K + k0 + ch0], &Bs[buf][ r0       * 32 + ch0]);
    copy_b128(&Bt[(size_t)(n0 + r0 + 64) * K + k0 + ch0], &Bs[buf][(r0 + 64) * 32 + ch0]);
  };

  stage(0, 0);
  int buf = 0;
  for (int k0 = 0; k0 < K; k0 += 32, buf ^= 1) {
    if (k0 + 32 < K) { stage(buf ^ 1, k0 + 32); async_wait<4>(); }
    else             { async_wait<0>(); }
    __syncthreads();                       // current buffer visible to all waves
    FragAB af[2], bfr[4];
    #pragma unroll
    for (int i = 0; i < 2; ++i) {          // A frags (16x32, row-per-lane)
      int r = wm * 32 + i * 16 + lm;
      af[i].q[0] = *(const uint4*)&As[buf][r * 32 + g * 8];
      af[i].q[1] = *(const uint4*)&As[buf][r * 32 + 16 + g * 8];
    }
    #pragma unroll
    for (int j = 0; j < 4; ++j) {          // B frags (32x16, K-contiguous)
      int r = wn * 64 + j * 16 + lm;
      bfr[j].q[0] = *(const uint4*)&Bs[buf][r * 32 + g * 8];
      bfr[j].q[1] = *(const uint4*)&Bs[buf][r * 32 + 16 + g * 8];
    }
    #pragma unroll
    for (int i = 0; i < 2; ++i)
      #pragma unroll
      for (int j = 0; j < 4; ++j)
        acc[i][j] = __builtin_amdgcn_wmma_f32_16x16x32_bf16(
            false, af[i].v, false, bfr[j].v, (short)0, acc[i][j], false, false);
    __syncthreads();                       // reads done before next-buf refill
  }

  #pragma unroll
  for (int i = 0; i < 2; ++i)
    #pragma unroll
    for (int j = 0; j < 4; ++j) {
      int col = n0 + wn * 64 + j * 16 + lm;
      float bv = bias[col];
      #pragma unroll
      for (int v = 0; v < 8; ++v) {
        int row = m0 + wm * 32 + i * 16 + g * 8 + v;
        float y = acc[i][j][v] + bv;
        if (RELU) y = fmaxf(y, 0.f);
        if (RES)  y += res[(size_t)row * N + col];
        if (OUT_F)  outF[(size_t)row * N + col] = y;
        if (OUT_BF) outB[(size_t)row * N + col] = f2bf(y);
      }
    }
}

// ---------------------------------------------------------------------------
// Flash causal GQA attention. qkv bf16 [B*T][2048] (q | k(512) | v(512)).
// Grid (T/128, H, B); block 256 = 8 waves; wave owns 16 q rows.
// GQA elementwise repeat: head hh dim d maps to kv feature hh*32 + d/2.
// ---------------------------------------------------------------------------
__global__ __launch_bounds__(256) void flash_gqa(
    const u16* __restrict__ qkv, u16* __restrict__ attn) {
  const int qt = blockIdx.x, hh = blockIdx.y, b = blockIdx.z;
  const int qbase = qt * 128;
  __shared__ __align__(16) u16 sK[128 * 64];   // Q staging (128x64), then K tile (64x64)
  __shared__ __align__(16) u16 sV[64 * 64];    // V^T tile [hd][key]
  __shared__ __align__(16) u16 sP[8 * 16 * 64]; // per-wave P (16x64) re-layout buffer
  const int t = threadIdx.x, w = t >> 5, l = t & 31, g = l >> 4, lm = l & 15;
  const size_t bT = (size_t)b * 1024;

  #pragma unroll
  for (int i = 0; i < 4; ++i) {                // stage Q tile 128x64 (async)
    int c = t + i * 256; int r = c >> 3, ch = (c & 7) * 8;
    copy_b128(&qkv[(bT + qbase + r) * 2048 + hh * 64 + ch], &sK[r * 64 + ch]);
  }
  async_wait<0>();
  __syncthreads();
  FragAB qf[2];
  #pragma unroll
  for (int ks = 0; ks < 2; ++ks) {             // Q A-frags for both hd k-steps
    int r = w * 16 + lm;
    qf[ks].q[0] = *(const uint4*)&sK[r * 64 + ks * 32 + g * 8];
    qf[ks].q[1] = *(const uint4*)&sK[r * 64 + ks * 32 + 16 + g * 8];
  }

  v8f O[4]; float Mr[8], Lr[8];
  #pragma unroll
  for (int j = 0; j < 4; ++j) O[j] = vzero8();
  #pragma unroll
  for (int v = 0; v < 8; ++v) { Mr[v] = -__builtin_inff(); Lr[v] = 0.f; }

  const int ktmax = 2 * qt + 1;                // 64-key tiles up to the diagonal
  for (int kt = 0; kt <= ktmax; ++kt) {
    const int kbase = kt * 64;
    __syncthreads();                           // prev reads of sK/sV/sP done
    #pragma unroll
    for (int i = 0; i < 8; ++i) {              // stage K (64x64) and V^T (64x64)
      int c = t + i * 256; int r = c >> 5, e = c & 31;
      unsigned kv = qkv[(bT + kbase + r) * 2048 + 1024 + hh * 32 + e];
      *(unsigned*)&sK[r * 64 + 2 * e] = kv | (kv << 16);  // GQA dup, packed b32
      u16 vv = qkv[(bT + kbase + r) * 2048 + 1536 + hh * 32 + e];
      sV[(2 * e) * 64 + r]     = vv;
      sV[(2 * e + 1) * 64 + r] = vv;
    }
    __syncthreads();

    v8f S[4];
    #pragma unroll
    for (int nt = 0; nt < 4; ++nt) {           // S = Q @ K^T (16x64 per wave)
      S[nt] = vzero8();
      #pragma unroll
      for (int ks = 0; ks < 2; ++ks) {
        FragAB kf;
        int r = nt * 16 + lm;
        kf.q[0] = *(const uint4*)&sK[r * 64 + ks * 32 + g * 8];
        kf.q[1] = *(const uint4*)&sK[r * 64 + ks * 32 + 16 + g * 8];
        S[nt] = __builtin_amdgcn_wmma_f32_16x16x32_bf16(
            false, qf[ks].v, false, kf.v, (short)0, S[nt], false, false);
      }
    }
    #pragma unroll
    for (int nt = 0; nt < 4; ++nt)             // scale 1/sqrt(64) + causal mask
      #pragma unroll
      for (int v = 0; v < 8; ++v) {
        int qi = qbase + w * 16 + g * 8 + v;
        int kj = kbase + nt * 16 + lm;
        float sc = S[nt][v] * 0.125f;
        S[nt][v] = (kj <= qi) ? sc : -__builtin_inff();
      }

    float alpha[8], psum[8];
    #pragma unroll
    for (int v = 0; v < 8; ++v) {              // row max across tiles + 16 lanes
      float m = fmaxf(fmaxf(S[0][v], S[1][v]), fmaxf(S[2][v], S[3][v]));
      #pragma unroll
      for (int off = 1; off < 16; off <<= 1) m = fmaxf(m, __shfl_xor(m, off, 32));
      float mn = fmaxf(Mr[v], m);
      alpha[v] = __expf(Mr[v] - mn);
      Mr[v] = mn; psum[v] = 0.f;
    }
    #pragma unroll
    for (int nt = 0; nt < 4; ++nt)             // P = exp(S - m), spill to LDS
      #pragma unroll
      for (int v = 0; v < 8; ++v) {
        float p = __expf(S[nt][v] - Mr[v]);
        sP[(w * 16 + g * 8 + v) * 64 + nt * 16 + lm] = f2bf(p);
        psum[v] += p;
      }
    #pragma unroll
    for (int v = 0; v < 8; ++v) {
      #pragma unroll
      for (int off = 1; off < 16; off <<= 1) psum[v] += __shfl_xor(psum[v], off, 32);
      Lr[v] = Lr[v] * alpha[v] + psum[v];
    }
    #pragma unroll
    for (int j = 0; j < 4; ++j)
      #pragma unroll
      for (int v = 0; v < 8; ++v) O[j][v] *= alpha[v];
    __syncthreads();                           // sP visible as A-fragments

    #pragma unroll
    for (int ks = 0; ks < 2; ++ks) {           // O += P @ V
      FragAB pf;
      pf.q[0] = *(const uint4*)&sP[(w * 16 + lm) * 64 + ks * 32 + g * 8];
      pf.q[1] = *(const uint4*)&sP[(w * 16 + lm) * 64 + ks * 32 + 16 + g * 8];
      #pragma unroll
      for (int j = 0; j < 4; ++j) {
        FragAB vf;
        int r = j * 16 + lm;
        vf.q[0] = *(const uint4*)&sV[r * 64 + ks * 32 + g * 8];
        vf.q[1] = *(const uint4*)&sV[r * 64 + ks * 32 + 16 + g * 8];
        O[j] = __builtin_amdgcn_wmma_f32_16x16x32_bf16(
            false, pf.v, false, vf.v, (short)0, O[j], false, false);
      }
    }
  }

  #pragma unroll
  for (int j = 0; j < 4; ++j)                  // O / l -> attn bf16 [B*T][1024]
    #pragma unroll
    for (int v = 0; v < 8; ++v) {
      int row = qbase + w * 16 + g * 8 + v;
      int col = hh * 64 + j * 16 + lm;
      attn[(bT + row) * 1024 + col] = f2bf(O[j][v] / Lr[v]);
    }
}

// ---------------------------------------------------------------------------
// Host-side launch. Workspace layout (byte offsets, all 16B aligned):
//   ln      bf16 [8192][1024]               @ 0          (16.78 MB)
//   WqkvT   bf16 [2048][1024]               @ 16777216
//   WoT     bf16 [1024][1024]               @ 20971520
//   W1T     bf16 [4096][1024]               @ 23068672
//   W2T     bf16 [1024][4096]               @ 31457280
//   qkv     bf16 [8192][2048]  (aliases h)  @ 39845888   (33.55 MB)
//   h       f32  [8192][1024]  (aliases qkv)@ 39845888
//   attn    bf16 [8192][1024]  (aliases mid)@ 73400320
//   mid     bf16 [8192][4096]  (aliases attn)@ 73400320  (67.11 MB)
// Total: ~140.5 MB.
// ---------------------------------------------------------------------------
extern "C" void kernel_launch(void* const* d_in, const int* in_sizes, int n_in,
                              void* d_out, int out_size, void* d_ws, size_t ws_size,
                              hipStream_t stream) {
  (void)in_sizes; (void)n_in; (void)out_size; (void)ws_size;
  const float* x     = (const float*)d_in[0];
  const float* ln1_g = (const float*)d_in[1];
  const float* ln1_b = (const float*)d_in[2];
  const float* Wqkv  = (const float*)d_in[3];
  const float* bqkv  = (const float*)d_in[4];
  const float* Wo    = (const float*)d_in[5];
  const float* bo    = (const float*)d_in[6];
  const float* ln2_g = (const float*)d_in[7];
  const float* ln2_b = (const float*)d_in[8];
  const float* W1    = (const float*)d_in[9];
  const float* b1    = (const float*)d_in[10];
  const float* W2    = (const float*)d_in[11];
  const float* b2    = (const float*)d_in[12];

  char* ws = (char*)d_ws;
  u16*   lnA   = (u16*)  (ws + 0);
  u16*   WqkvT = (u16*)  (ws + 16777216);
  u16*   WoT   = (u16*)  (ws + 20971520);
  u16*   W1T   = (u16*)  (ws + 23068672);
  u16*   W2T   = (u16*)  (ws + 31457280);
  u16*   qkvB  = (u16*)  (ws + 39845888);
  float* hF    = (float*)(ws + 39845888);   // alias qkv (dead after flash)
  u16*   attnB = (u16*)  (ws + 73400320);
  u16*   midB  = (u16*)  (ws + 73400320);   // alias attn (dead after Wo gemm)

  // 1) weight transpose+convert (one-time per call; bandwidth trivial)
  transpose_cvt<<<(1024 * 2048) / 256, 256, 0, stream>>>(Wqkv, WqkvT, 1024, 2048);
  transpose_cvt<<<(1024 * 1024) / 256, 256, 0, stream>>>(Wo,   WoT,   1024, 1024);
  transpose_cvt<<<(1024 * 4096) / 256, 256, 0, stream>>>(W1,   W1T,   1024, 4096);
  transpose_cvt<<<(4096 * 1024) / 256, 256, 0, stream>>>(W2,   W2T,   4096, 1024);

  // 2) LN1
  layernorm_bf16<<<8192, 256, 0, stream>>>(x, ln1_g, ln1_b, lnA);

  // 3) QKV = ln1 @ Wqkv + bqkv  (bf16 out)
  gemm_bf16<true, false, false, false><<<dim3(2048 / 128, 8192 / 128), 256, 0, stream>>>(
      lnA, WqkvT, bqkv, nullptr, qkvB, nullptr, 8192, 2048, 1024);

  // 4) flash causal GQA attention
  flash_gqa<<<dim3(1024 / 128, 16, 8), 256, 0, stream>>>(qkvB, attnB);

  // 5) h = attn @ Wo + bo + x   (f32 out)
  gemm_bf16<false, true, false, true><<<dim3(1024 / 128, 8192 / 128), 256, 0, stream>>>(
      attnB, WoT, bo, x, nullptr, hF, 8192, 1024, 1024);

  // 6) LN2(h)
  layernorm_bf16<<<8192, 256, 0, stream>>>(hF, ln2_g, ln2_b, lnA);

  // 7) mid = relu(ln2 @ W1 + b1)  (bf16 out)
  gemm_bf16<true, false, true, false><<<dim3(4096 / 128, 8192 / 128), 256, 0, stream>>>(
      lnA, W1T, b1, nullptr, midB, nullptr, 8192, 4096, 1024);

  // 8) out = mid @ W2 + b2 + x   (f32 out)
  gemm_bf16<false, true, false, true><<<dim3(1024 / 128, 8192 / 128), 256, 0, stream>>>(
      midB, W2T, b2, x, nullptr, (float*)d_out, 8192, 1024, 4096);
}
